// MetaS4_70506183131278
// MI455X (gfx1250) — matrix operand order repaired and checked by reading, not compiled, not run on hardware
//
#include <hip/hip_runtime.h>
#include <math.h>

#define B_SZ 1024
#define H_SZ 1024
#define N_SZ 32

typedef __attribute__((ext_vector_type(16))) __bf16        v16bf;
typedef __attribute__((ext_vector_type(8)))  float         v8f;
typedef __attribute__((ext_vector_type(4)))  unsigned int  u32x4;
typedef __attribute__((ext_vector_type(8)))  int           i32x8;
typedef __attribute__((ext_vector_type(4)))  int           i32x4;

__device__ inline void bf_split(float x, __bf16& hi, __bf16& lo) {
    hi = (__bf16)x;
    lo = (__bf16)(x - (float)hi);
}

// ---------------------------------------------------------------------------
// Kernel 0: one-shot fp32 -> bf16 (hi, lo) split of W.  16MB of traffic,
// amortized over the whole GEMM (replaces 64x-redundant in-loop conversion).
// ---------------------------------------------------------------------------
__global__ __launch_bounds__(256) void cvt_split_kernel(
    const float* __restrict__ src, __bf16* __restrict__ hi,
    __bf16* __restrict__ lo)
{
    const int i = blockIdx.x * blockDim.x + threadIdx.x;   // one per 8 floats
    const float4 v0 = ((const float4*)src)[2 * i];
    const float4 v1 = ((const float4*)src)[2 * i + 1];
    const float xs[8] = {v0.x, v0.y, v0.z, v0.w, v1.x, v1.y, v1.z, v1.w};
    __bf16 h[8], l[8];
    #pragma unroll
    for (int j = 0; j < 8; ++j) bf_split(xs[j], h[j], l[j]);
    __builtin_memcpy(hi + 8 * (long)i, h, 16);
    __builtin_memcpy(lo + 8 * (long)i, l, 16);
}

// ---------------------------------------------------------------------------
// Kernel 1: complex diagonal state update + readout + skip + exact GELU.
// One wave32 per (b,h); lane = n (N == 32 == warpSize). All global traffic is
// 128B-coalesced per wave. Produces ns_re/ns_im (to d_out) and bf16 hi/lo y
// (to ws).  Purely HBM-bandwidth bound: ~520MB @ 23.3 TB/s.
// ---------------------------------------------------------------------------
__global__ __launch_bounds__(256) void s4_state_kernel(
    const float* __restrict__ u,      // [B,H]
    const float* __restrict__ s_re,   // [B,H,N]
    const float* __restrict__ s_im,   // [B,H,N]
    const float* __restrict__ dA_re,  // [H,N]
    const float* __restrict__ dA_im,  // [H,N]
    const float* __restrict__ dB_re,  // [H,N]
    const float* __restrict__ dB_im,  // [H,N]
    const float* __restrict__ C_re,   // [H,N]
    const float* __restrict__ C_im,   // [H,N]
    const float* __restrict__ D,      // [H]
    float*  __restrict__ ns_re,       // [B,H,N] out
    float*  __restrict__ ns_im,       // [B,H,N] out
    __bf16* __restrict__ y_hi,        // [B,H] bf16 staging (ws)
    __bf16* __restrict__ y_lo)        // [B,H] bf16 staging (ws)
{
    const int  lane = threadIdx.x & 31;
    const int  wib  = threadIdx.x >> 5;
    const long pair = (long)blockIdx.x * (blockDim.x >> 5) + wib;   // b*H + h
    const int  h    = (int)(pair & (H_SZ - 1));
    const long base = pair * N_SZ + lane;
    const int  hn   = h * N_SZ + lane;

    const float sr = s_re[base];
    const float si = s_im[base];
    const float ar = dA_re[hn], ai = dA_im[hn];
    const float br = dB_re[hn], bi = dB_im[hn];
    const float uu = u[pair];

    const float nr = fmaf(ar, sr, fmaf(-ai, si, br * uu));
    const float ni = fmaf(ar, si, fmaf( ai, sr, bi * uu));
    ns_re[base] = nr;
    ns_im[base] = ni;

    float c = fmaf(C_re[hn], nr, -C_im[hn] * ni);
    #pragma unroll
    for (int off = 16; off > 0; off >>= 1)
        c += __shfl_xor(c, off, 32);

    if (lane == 0) {
        float y = 2.0f * c + uu * D[h];
        y = 0.5f * y * (1.0f + erff(y * 0.70710678118654752f));  // exact GELU
        __bf16 hi, lo;
        bf_split(y, hi, lo);
        y_hi[pair] = hi;
        y_lo[pair] = lo;
    }
}

// ---------------------------------------------------------------------------
// TDM helper: one descriptor moves a [tile_rows x 32] bf16 tile with global
// row stride 1024 elements into LDS, with HW padding of 4 DWORDs after every
// 16 DWORDs (one 64B row) -> LDS row stride 40 bf16 = 20 banks, conflict-free
// fragment reads.  D# layout per CDNA5 ISA.
// ---------------------------------------------------------------------------
__device__ inline void tdm_load_tile(unsigned lds_off, const __bf16* gptr,
                                     int tile_rows)
{
    const unsigned long long ga = (unsigned long long)(uintptr_t)gptr;
    u32x4 g0;
    g0[0] = 1u;                                        // count=1 (valid user D#)
    g0[1] = lds_off;                                   // lds_addr (bytes)
    g0[2] = (unsigned)(ga & 0xFFFFFFFFu);              // global_addr[31:0]
    g0[3] = (unsigned)((ga >> 32) & 0x01FFFFFFu)       // global_addr[56:32]
          | (2u << 30);                                // type=2 ("image")
    i32x8 g1;
    g1[0] = (1 << 16)                                  // data_size = 2 bytes
          | (1 << 20)                                  // pad_enable
          | (3 << 22)                                  // pad_interval = 16 DW
          | (3 << 25);                                 // pad_amount   = 4 DW
    g1[1] = (int)((H_SZ & 0xFFFF) << 16);              // tensor_dim0[15:0]
    g1[2] = (int)((H_SZ & 0xFFFF) << 16);              // dim0[31:16]=0 | dim1[15:0]
    g1[3] = (int)(32u << 16);                          // dim1[31:16]=0 | tile_dim0=32
    g1[4] = tile_rows;                                 // tile_dim1 | tile_dim2=0
    g1[5] = H_SZ;                                      // tensor_dim0_stride lo32
    g1[6] = 0;                                         // stride hi | dim1_stride lo
    g1[7] = 0;
    const i32x4 z4 = {0, 0, 0, 0};                     // 2-D tensor: groups 2/3 unused
    const i32x8 z8 = {0, 0, 0, 0, 0, 0, 0, 0};
    __builtin_amdgcn_tensor_load_to_lds(g0, g1, z4, z4, z8, 0);
}

// ---------------------------------------------------------------------------
// Kernel 2: fused Linear(H->2H) + GLU via WMMA bf16 hi/lo split
// (z ~= Ah*Bh + Ah*Bl + Al*Bh, fp32 accumulate -> near-fp32 accuracy).
// Double-buffered LDS fed by the Tensor Data Mover (TENSORcnt): the inner
// loop is pure data-path (12 ds_load_b128 + 6 WMMA + 6 TDM issues), no VALU.
// Block = 256 threads (8 waves); tile = 64 rows x 32 linear cols + the paired
// gate cols (j+1024) so GLU fuses into the epilogue.
// ---------------------------------------------------------------------------
#define LROW 40   // LDS bf16 row stride (32 + 8 pad elements, matches TDM pad)

__global__ __launch_bounds__(256) void s4_glu_gemm_kernel(
    const __bf16* __restrict__ y_hi,   // [B,H] (ws)
    const __bf16* __restrict__ y_lo,   // [B,H] (ws)
    const __bf16* __restrict__ W_hi,   // [2H,H] (ws)
    const __bf16* __restrict__ W_lo,   // [2H,H] (ws)
    const float*  __restrict__ bias,   // [2H]
    float* __restrict__ out)           // [B,H]
{
    __shared__ __bf16 Ah[2][64][LROW];   // y tile hi
    __shared__ __bf16 Al[2][64][LROW];   // y tile lo
    __shared__ __bf16 Blh[2][32][LROW];  // W linear hi
    __shared__ __bf16 Bll[2][32][LROW];  // W linear lo
    __shared__ __bf16 Bgh[2][32][LROW];  // W gate hi
    __shared__ __bf16 Bgl[2][32][LROW];  // W gate lo

    const int t     = threadIdx.x;
    const int lane  = t & 31;
    const int wave  = t >> 5;
    const int m_sub = wave & 3;   // 16-row subtile
    const int n_sub = wave >> 2;  // 16-col subtile

    const int mblk = blockIdx.x & 15;   // 16 tiles of 64 rows
    const int nblk = blockIdx.x >> 4;   // 32 tiles of 32 cols (first half)
    const int m0   = mblk * 64;
    const int n0   = nblk * 32;

    const bool issuer = (wave == 0);    // TDM: one wave issues descriptors

    const __bf16* ah_base  = y_hi + (long)m0 * H_SZ;
    const __bf16* al_base  = y_lo + (long)m0 * H_SZ;
    const __bf16* blh_base = W_hi + (long)n0 * H_SZ;
    const __bf16* bll_base = W_lo + (long)n0 * H_SZ;
    const __bf16* bgh_base = W_hi + (long)(n0 + H_SZ) * H_SZ;
    const __bf16* bgl_base = W_lo + (long)(n0 + H_SZ) * H_SZ;

    #define ISSUE6(buf, k0)                                                    \
        tdm_load_tile((unsigned)(uintptr_t)&Ah[buf][0][0],  ah_base  + (k0), 64); \
        tdm_load_tile((unsigned)(uintptr_t)&Al[buf][0][0],  al_base  + (k0), 64); \
        tdm_load_tile((unsigned)(uintptr_t)&Blh[buf][0][0], blh_base + (k0), 32); \
        tdm_load_tile((unsigned)(uintptr_t)&Bll[buf][0][0], bll_base + (k0), 32); \
        tdm_load_tile((unsigned)(uintptr_t)&Bgh[buf][0][0], bgh_base + (k0), 32); \
        tdm_load_tile((unsigned)(uintptr_t)&Bgl[buf][0][0], bgl_base + (k0), 32)

    if (issuer) { ISSUE6(0, 0); }   // prologue: fill buffer 0

    v8f acc_lin  = {};
    v8f acc_gate = {};

    const int fr = lane & 15;
    const int kb = (lane >> 4) * 8;    // unused: A loads also use kh, lanes<16 -> K{0..15}
    const int kh = (lane >> 4) * 16;   // B K-half: lanes<16 -> K{0..15}

    for (int step = 0; step < H_SZ / 32; ++step) {
        const int p  = step & 1;
        const int kn = (step + 1) * 32;

        if (issuer) {
            if (kn < H_SZ) {   // issue next buffer, then wait for current
                ISSUE6(p ^ 1, kn);
                __builtin_amdgcn_s_wait_tensorcnt((short)6);  // prev 6 complete
            } else {
                __builtin_amdgcn_s_wait_tensorcnt((short)0);
            }
        }
        __syncthreads();   // current buffer visible to all waves

        // ---- fragments straight from bf16 LDS (no conversion in loop) ----
        v16bf a_hi, a_lo, bl_h, bl_l, bg_h, bg_l;
        {
            const __bf16* ahr = &Ah[p][m_sub * 16 + fr][0];
            const __bf16* alr = &Al[p][m_sub * 16 + fr][0];
            __builtin_memcpy(&a_hi, ahr + kh, 32);
            __builtin_memcpy(&a_lo, alr + kh, 32);
            __builtin_memcpy(&bl_h, &Blh[p][n_sub * 16 + fr][0] + kh, 32);
            __builtin_memcpy(&bl_l, &Bll[p][n_sub * 16 + fr][0] + kh, 32);
            __builtin_memcpy(&bg_h, &Bgh[p][n_sub * 16 + fr][0] + kh, 32);
            __builtin_memcpy(&bg_l, &Bgl[p][n_sub * 16 + fr][0] + kh, 32);
        }

        acc_lin = __builtin_amdgcn_wmma_f32_16x16x32_bf16(
            false, a_hi, false, bl_h, (short)0, acc_lin, false, false);
        acc_lin = __builtin_amdgcn_wmma_f32_16x16x32_bf16(
            false, a_hi, false, bl_l, (short)0, acc_lin, false, false);
        acc_lin = __builtin_amdgcn_wmma_f32_16x16x32_bf16(
            false, a_lo, false, bl_h, (short)0, acc_lin, false, false);

        acc_gate = __builtin_amdgcn_wmma_f32_16x16x32_bf16(
            false, a_hi, false, bg_h, (short)0, acc_gate, false, false);
        acc_gate = __builtin_amdgcn_wmma_f32_16x16x32_bf16(
            false, a_hi, false, bg_l, (short)0, acc_gate, false, false);
        acc_gate = __builtin_amdgcn_wmma_f32_16x16x32_bf16(
            false, a_lo, false, bg_h, (short)0, acc_gate, false, false);

        __syncthreads();   // reads done before TDM refills this buffer
    }
    #undef ISSUE6

    // ---- fused bias + GLU epilogue ----
    // C/D layout: VGPR r, lanes 0-15 -> row r, lanes 16-31 -> row r+8.
    const int col = n0 + n_sub * 16 + (lane & 15);
    const float bl = bias[col];
    const float bg = bias[col + H_SZ];
    const int rbase = m0 + m_sub * 16 + ((lane >> 4) * 8);
    #pragma unroll
    for (int r = 0; r < 8; ++r) {
        const float zl = acc_lin[r] + bl;
        const float zg = acc_gate[r] + bg;
        out[(long)(rbase + r) * H_SZ + col] = zl / (1.0f + __expf(-zg));
    }
}

// ---------------------------------------------------------------------------
extern "C" void kernel_launch(void* const* d_in, const int* in_sizes, int n_in,
                              void* d_out, int out_size, void* d_ws, size_t ws_size,
                              hipStream_t stream) {
    const float* u     = (const float*)d_in[0];
    const float* s_re  = (const float*)d_in[1];
    const float* s_im  = (const float*)d_in[2];
    const float* dA_re = (const float*)d_in[3];
    const float* dA_im = (const float*)d_in[4];
    const float* dB_re = (const float*)d_in[5];
    const float* dB_im = (const float*)d_in[6];
    const float* C_re  = (const float*)d_in[7];
    const float* C_im  = (const float*)d_in[8];
    const float* D     = (const float*)d_in[9];
    const float* W     = (const float*)d_in[10];
    const float* bias  = (const float*)d_in[11];

    float* out   = (float*)d_out;                       // y_out [B,H]
    float* ns_re = out + (long)B_SZ * H_SZ;             // [B,H,N]
    float* ns_im = ns_re + (long)B_SZ * H_SZ * N_SZ;    // [B,H,N]

    // ws layout (12MB total): y_hi 2MB | y_lo 2MB | W_hi 4MB | W_lo 4MB
    char*   ws   = (char*)d_ws;
    __bf16* y_hi = (__bf16*)(ws);
    __bf16* y_lo = (__bf16*)(ws + (2u << 20));
    __bf16* W_hi = (__bf16*)(ws + (4u << 20));
    __bf16* W_lo = (__bf16*)(ws + (8u << 20));

    // Kernel 0: split W (2M elements, 8 per thread).
    cvt_split_kernel<<<(2 * H_SZ * H_SZ) / (8 * 256), 256, 0, stream>>>(
        W, W_hi, W_lo);

    // Kernel 1: 1M (b,h) waves, 8 waves per 256-thread block.
    const int blocks1 = (B_SZ * H_SZ) / 8;
    s4_state_kernel<<<blocks1, 256, 0, stream>>>(
        u, s_re, s_im, dA_re, dA_im, dB_re, dB_im, C_re, C_im, D,
        ns_re, ns_im, y_hi, y_lo);

    // Kernel 2: 16 M-tiles x 32 N-tiles = 512 blocks of 256 threads.
    s4_glu_gemm_kernel<<<16 * 32, 256, 0, stream>>>(
        y_hi, y_lo, W_hi, W_lo, bias, out);
}
